// RNN_Hierarchical_17686675325350
// MI455X (gfx1250) — matrix operand order repaired and checked by reading, not compiled, not run on hardware
//
#include <hip/hip_runtime.h>

// ---------------------------------------------------------------------------
// Hierarchical RNN (D=3, T=784, B=256, H=512, I=28, C=2) for MI455X (gfx1250).
//
// Roofline: 526 GFLOP of recurrent GEMMs, weights (~5MB) + state (~1.5MB)
// live in the 192MB L2 -> compute/latency bound, not HBM bound. Use
// v_wmma_f32_16x16x32_bf16 (bf16 inputs, f32 accumulate) for 8x density over
// the f32 WMMA shape. Sequential dependence handled with a skewed (t+d)
// pipeline: 786 graph launches instead of 2352, 3 depths per launch.
// ---------------------------------------------------------------------------

typedef __bf16 bf16_t;
typedef __attribute__((ext_vector_type(16))) __bf16 v16bf;
typedef __attribute__((ext_vector_type(8)))  __bf16 v8bf;
typedef __attribute__((ext_vector_type(8)))  float  v8f;

#define RD 3
#define RT 784
#define RB 256
#define RH 512
#define RI 28
#define RIP 32      // input K padded to one 32-wide WMMA chunk
#define RC 2
#define NEG_SLOPE 0.01f

// ---- workspace layout (bytes) ----
#define OFF_WT_HH 0u
#define SZ_WT_HH  (RD*RH*RH*2u)                 // 1,572,864
#define OFF_WT_FF (OFF_WT_HH + SZ_WT_HH)
#define SZ_WT_FF  (RD*RH*RH*2u)
#define OFF_WT_IN (OFF_WT_FF + SZ_WT_FF)
#define SZ_WT_IN  (RD*RIP*RH*2u)                // 98,304
#define OFF_CBIAS (OFF_WT_IN + SZ_WT_IN)
#define SZ_CBIAS  (RD*RH*4u)
#define OFF_SLOTF (OFF_CBIAS + SZ_CBIAS)
#define SZ_SLOTF  (2u*RD*RB*RH*4u)              // 3,145,728
#define OFF_SLOTB (OFF_SLOTF + SZ_SLOTF)
#define SZ_SLOTB  (2u*RD*RB*RH*2u)
#define OFF_DATA  (OFF_SLOTB + SZ_SLOTB)
#define SZ_DATA   ((unsigned)RT*RB*RIP*2u)      // 12,845,056  (total ~20.8 MB)

// ---------------------------------------------------------------------------
// Fragment loaders per CDNA5 16-bit WMMA VGPR layouts (ISA 7.12.2).
// A (16x32, MxK): lanes 0-15 row M=lane, K in {k0..k0+7, k0+16..k0+23};
//                 lanes 16-31 row M=lane-16, K in {k0+8..15, k0+24..31}.
//                 -> two 16B contiguous chunks per lane.
// B (32x16, KxN): column N=lane%16; K order same as A (lanes 0-15:
//                 k0..k0+7, k0+16..k0+23; lanes 16-31: k0+8..15, k0+24..31),
//                 gathered from Wt[k][g] (W stored pre-transposed).
// ---------------------------------------------------------------------------
__device__ __forceinline__ v16bf load_a_frag(const bf16_t* base, int stride,
                                             int m0, int k0, int lane) {
  const int mr = m0 + (lane & 15);
  const bf16_t* p = base + (size_t)mr * stride + k0 + ((lane >> 4) << 3);
  union { v16bf v; v8bf h[2]; } u;
  u.h[0] = *(const v8bf*)p;
  u.h[1] = *(const v8bf*)(p + 16);
  return u.v;
}

__device__ __forceinline__ v16bf load_b_frag(const bf16_t* base, int k0,
                                             int n0, int lane) {
  const int n = n0 + (lane & 15);
  const int kb = k0 + ((lane >> 4) << 3);
  v16bf v;
#pragma unroll
  for (int s = 0; s < 8; ++s) {
    v[s] = base[(size_t)(kb + s) * RH + n];
    v[s + 8] = base[(size_t)(kb + 16 + s) * RH + n];
  }
  return v;
}

// ---------------------------------------------------------------------------
// One pipeline stage: for all depths d with t = iter - d in [0,T), compute
//   pre  = h_prev(d) @ W_hh(d)^T + [d>0] h_new(d-1) @ W_ff(d)^T
//        + x_t @ W_in(d)^T + (b_in+b_hh+b_ff)
//   h_new = alpha*h_prev + leaky_relu(pre)/tau
// Wave tile: 16(batch) x 64(hidden) -> 4x v_wmma_f32_16x16x32_bf16 per chunk.
// Block: 8 waves as 2(M) x 4(N). Grid: (H/256, B/32, D) = (2, 8, 3).
// ---------------------------------------------------------------------------
__global__ __launch_bounds__(256) void rnn_step_kernel(
    int iter,
    const float*  __restrict__ hprevF, const bf16_t* __restrict__ hprevB,
    float*        __restrict__ hcurF,  bf16_t*       __restrict__ hcurB,
    const bf16_t* __restrict__ wtHH,   const bf16_t* __restrict__ wtFF,
    const bf16_t* __restrict__ wtIN,   const float*  __restrict__ cbias,
    const bf16_t* __restrict__ dataB,  const float*  __restrict__ taus,
    float*        __restrict__ outH)
{
  const int d = blockIdx.z;
  const int t = iter - d;
  if (t < 0 || t >= RT) return;

  const int lane = threadIdx.x & 31;
  const int wave = threadIdx.x >> 5;
  const int m0 = blockIdx.y * 32 + (wave & 1) * 16;        // batch rows
  const int n0 = blockIdx.x * 256 + (wave >> 1) * 64;      // hidden cols

  v8f acc[4] = {};

  // Phase 0: recurrent GEMM, K = 512 over h_prev(d)
  {
    const bf16_t* hp = hprevB + (size_t)d * RB * RH;
    const bf16_t* w  = wtHH   + (size_t)d * RH * RH;
#pragma unroll 4
    for (int k0 = 0; k0 < RH; k0 += 32) {
      v16bf a = load_a_frag(hp, RH, m0, k0, lane);
#pragma unroll
      for (int j = 0; j < 4; ++j) {
        v16bf b = load_b_frag(w, k0, n0 + 16 * j, lane);
        acc[j] = __builtin_amdgcn_wmma_f32_16x16x32_bf16(
            false, a, false, b, (short)0, acc[j], false, false);
      }
    }
  }

  // Phase 1: feed-forward GEMM from lower depth's fresh state (prev slot)
  if (d > 0) {
    const bf16_t* hl = hprevB + (size_t)(d - 1) * RB * RH;
    const bf16_t* w  = wtFF   + (size_t)d * RH * RH;
#pragma unroll 4
    for (int k0 = 0; k0 < RH; k0 += 32) {
      v16bf a = load_a_frag(hl, RH, m0, k0, lane);
#pragma unroll
      for (int j = 0; j < 4; ++j) {
        v16bf b = load_b_frag(w, k0, n0 + 16 * j, lane);
        acc[j] = __builtin_amdgcn_wmma_f32_16x16x32_bf16(
            false, a, false, b, (short)0, acc[j], false, false);
      }
    }
  }

  // Phase 2: input projection, single K=32 chunk (K padded 28 -> 32)
  {
    const bf16_t* xd = dataB + (size_t)t * RB * RIP;
    const bf16_t* w  = wtIN  + (size_t)d * RIP * RH;
    v16bf a = load_a_frag(xd, RIP, m0, 0, lane);
#pragma unroll
    for (int j = 0; j < 4; ++j) {
      v16bf b = load_b_frag(w, 0, n0 + 16 * j, lane);
      acc[j] = __builtin_amdgcn_wmma_f32_16x16x32_bf16(
          false, a, false, b, (short)0, acc[j], false, false);
    }
  }

  // Epilogue: bias + leaky_relu + alpha blend; D layout per ISA:
  // VGPR e, lanes 0-15: M=e, N=lane; lanes 16-31: M=e+8, N=lane-16.
  const float tau  = taus[d];
  const float invt = 1.0f / tau;
  const float al   = 1.0f - invt;
  const float*  hpF = hprevF + (size_t)d * RB * RH;
  float*        hcF = hcurF  + (size_t)d * RB * RH;
  bf16_t*       hcB = hcurB  + (size_t)d * RB * RH;
  const int half = lane >> 4, nc = lane & 15;

#pragma unroll
  for (int j = 0; j < 4; ++j) {
    const int g  = n0 + 16 * j + nc;
    const float bb = cbias[d * RH + g];
#pragma unroll
    for (int e = 0; e < 8; ++e) {
      const int m = m0 + e + 8 * half;
      float pre = acc[j][e] + bb;
      float lr  = pre > 0.0f ? pre : NEG_SLOPE * pre;
      float hn  = al * hpF[(size_t)m * RH + g] + lr * invt;
      hcF[(size_t)m * RH + g] = hn;
      hcB[(size_t)m * RH + g] = (bf16_t)hn;
      if (t == RT - 1) outH[((size_t)d * RB + m) * RH + g] = hn;
    }
  }
}

// ---------------------------------------------------------------------------
// Prep kernels (one-time per call, all L2-hot afterwards)
// ---------------------------------------------------------------------------
__global__ void prep_wt_sq(const float* __restrict__ W, bf16_t* __restrict__ Wt,
                           int zeroDiag) {
  size_t idx = (size_t)blockIdx.x * blockDim.x + threadIdx.x;
  if (idx >= (size_t)RD * RH * RH) return;
  int g = (int)(idx % RH);
  int k = (int)((idx / RH) % RH);
  int d = (int)(idx / ((size_t)RH * RH));
  float v = W[((size_t)d * RH + g) * RH + k];     // Wt[d][k][g] = W[d][g][k]
  if (zeroDiag && g == k) v = 0.0f;
  Wt[idx] = (bf16_t)v;
}

__global__ void prep_wt_in(const float* __restrict__ Win, bf16_t* __restrict__ Wt) {
  size_t idx = (size_t)blockIdx.x * blockDim.x + threadIdx.x;
  if (idx >= (size_t)RD * RIP * RH) return;
  int h = (int)(idx % RH);
  int k = (int)((idx / RH) % RIP);
  int d = (int)(idx / ((size_t)RIP * RH));
  float v = (k < RI) ? Win[((size_t)d * RH + h) * RI + k] : 0.0f;
  Wt[idx] = (bf16_t)v;                             // Wt[d][k<32][h]
}

__global__ void prep_bias(const float* __restrict__ b_in,
                          const float* __restrict__ b_hh,
                          const float* __restrict__ b_ff,
                          float* __restrict__ cb) {
  int idx = blockIdx.x * blockDim.x + threadIdx.x;
  if (idx >= RD * RH) return;
  int d = idx / RH;
  cb[idx] = b_in[idx] + b_hh[idx] + (d > 0 ? b_ff[idx] : 0.0f);
}

__global__ void prep_h0(const float* __restrict__ h0,
                        float* __restrict__ s0f, float* __restrict__ s1f,
                        bf16_t* __restrict__ s0b, bf16_t* __restrict__ s1b) {
  size_t idx = (size_t)blockIdx.x * blockDim.x + threadIdx.x;
  if (idx >= (size_t)RD * RB * RH) return;
  float v = h0[idx];
  s0f[idx] = v; s1f[idx] = v;
  s0b[idx] = (bf16_t)v; s1b[idx] = (bf16_t)v;
}

__global__ void prep_data(const float* __restrict__ data, bf16_t* __restrict__ db) {
  size_t idx = (size_t)blockIdx.x * blockDim.x + threadIdx.x;
  if (idx >= (size_t)RT * RB * RIP) return;
  int k = (int)(idx % RIP);
  size_t tb = idx / RIP;
  db[idx] = (k < RI) ? (bf16_t)data[tb * RI + k] : (bf16_t)0.0f;
}

// ---------------------------------------------------------------------------
// Readout: out[d,b,c] = sum_h hfin[d,b,h]*W_fc[d,c,h] + b_fc[d,c]
// One wave per (d,b); lane-parallel over h with shuffle reduction (wave32).
// ---------------------------------------------------------------------------
__global__ __launch_bounds__(256) void readout_kernel(
    const float* __restrict__ hfin, const float* __restrict__ Wfc,
    const float* __restrict__ bfc, float* __restrict__ out) {
  int wid  = (int)((blockIdx.x * blockDim.x + threadIdx.x) >> 5);
  int lane = threadIdx.x & 31;
  if (wid >= RD * RB) return;
  int d = wid / RB;
  const float* hv = hfin + (size_t)wid * RH;
  const float* w0 = Wfc + ((size_t)d * RC + 0) * RH;
  const float* w1 = Wfc + ((size_t)d * RC + 1) * RH;
  float s0 = 0.0f, s1 = 0.0f;
  for (int h = lane; h < RH; h += 32) {
    float x = hv[h];
    s0 += x * w0[h];
    s1 += x * w1[h];
  }
  for (int m = 16; m >= 1; m >>= 1) {
    s0 += __shfl_xor(s0, m, 32);
    s1 += __shfl_xor(s1, m, 32);
  }
  if (lane == 0) {
    out[(size_t)wid * RC + 0] = s0 + bfc[d * RC + 0];
    out[(size_t)wid * RC + 1] = s1 + bfc[d * RC + 1];
  }
}

// ---------------------------------------------------------------------------
extern "C" void kernel_launch(void* const* d_in, const int* in_sizes, int n_in,
                              void* d_out, int out_size, void* d_ws, size_t ws_size,
                              hipStream_t stream) {
  const float* data = (const float*)d_in[0];
  const float* h0   = (const float*)d_in[1];
  const float* W_in = (const float*)d_in[2];
  const float* b_in = (const float*)d_in[3];
  const float* W_hh = (const float*)d_in[4];
  const float* b_hh = (const float*)d_in[5];
  const float* W_ff = (const float*)d_in[6];
  const float* b_ff = (const float*)d_in[7];
  const float* W_fc = (const float*)d_in[8];
  const float* b_fc = (const float*)d_in[9];
  const float* taus = (const float*)d_in[10];

  char* ws = (char*)d_ws;
  bf16_t* wt_hh = (bf16_t*)(ws + OFF_WT_HH);
  bf16_t* wt_ff = (bf16_t*)(ws + OFF_WT_FF);
  bf16_t* wt_in = (bf16_t*)(ws + OFF_WT_IN);
  float*  cbias = (float*) (ws + OFF_CBIAS);
  float*  slotF0 = (float*)(ws + OFF_SLOTF);
  float*  slotF1 = slotF0 + (size_t)RD * RB * RH;
  bf16_t* slotB0 = (bf16_t*)(ws + OFF_SLOTB);
  bf16_t* slotB1 = slotB0 + (size_t)RD * RB * RH;
  bf16_t* dataB  = (bf16_t*)(ws + OFF_DATA);

  float* slotF[2] = { slotF0, slotF1 };
  bf16_t* slotB[2] = { slotB0, slotB1 };

  // ---- prep ----
  prep_wt_sq<<<(RD * RH * RH + 255) / 256, 256, 0, stream>>>(W_hh, wt_hh, 1);
  prep_wt_sq<<<(RD * RH * RH + 255) / 256, 256, 0, stream>>>(W_ff, wt_ff, 0);
  prep_wt_in<<<(RD * RIP * RH + 255) / 256, 256, 0, stream>>>(W_in, wt_in);
  prep_bias<<<(RD * RH + 255) / 256, 256, 0, stream>>>(b_in, b_hh, b_ff, cbias);
  prep_h0<<<(RD * RB * RH + 255) / 256, 256, 0, stream>>>(h0, slotF0, slotF1,
                                                          slotB0, slotB1);
  prep_data<<<((size_t)RT * RB * RIP + 255) / 256, 256, 0, stream>>>(data, dataB);

  float* outH = (float*)d_out;                       // h_final [D,B,H]
  float* outC = (float*)d_out + (size_t)RD * RB * RH; // logits  [D,B,C]

  // ---- skewed (t+d) pipeline: 786 launches, 3 depths each ----
  dim3 grid(RH / 256, RB / 32, RD);   // (2, 8, 3) blocks, 8 waves/block
  for (int it = 0; it < RT + RD - 1; ++it) {
    const int cur = it & 1, prv = cur ^ 1;
    rnn_step_kernel<<<grid, 256, 0, stream>>>(
        it, slotF[prv], slotB[prv], slotF[cur], slotB[cur],
        wt_hh, wt_ff, wt_in, cbias, dataB, taus, outH);
  }

  // ---- readout heads ----
  readout_kernel<<<(RD * RB * 32 + 255) / 256, 256, 0, stream>>>(
      outH, W_fc, b_fc, outC);
}
